// Attention_38225208934700
// MI455X (gfx1250) — matrix operand order repaired and checked by reading, not compiled, not run on hardware
//
#include <hip/hip_runtime.h>
#include <math.h>

// Problem constants (from reference)
#define NB     16      // batch
#define NH     32      // query heads
#define NKVH   8       // kv heads
#define ND     128     // head dim
#define NCACHE 4096    // cache length
#define NREP   4       // NH / NKVH
#define NSPLIT 8       // sequence splits (flash-decode partials)
#define NWAVE  8       // waves per block (256 threads, wave32)
#define TILE   32      // seq positions per tile (two 16-wide WMMA subtiles)
#define PSZ    (NREP*ND + 2*NREP)   // floats per partial: acc[4][128], m[4], l[4]
#define INV_SQRT_D 0.08838834764831845f

typedef __attribute__((ext_vector_type(16))) _Float16 v16h;
typedef __attribute__((ext_vector_type(8)))  float    v8f;
typedef __attribute__((ext_vector_type(4)))  int      v4i;
typedef __attribute__((ext_vector_type(4)))  float    v4f;

__device__ __forceinline__ float grp16_max(float v) {
#pragma unroll
  for (int off = 1; off < 16; off <<= 1) v = fmaxf(v, __shfl_xor(v, off, 32));
  return v;
}
__device__ __forceinline__ float grp16_sum(float v) {
#pragma unroll
  for (int off = 1; off < 16; off <<= 1) v += __shfl_xor(v, off, 32);
  return v;
}

// Streams cached positions s in [0, P).  The new token at s == P is handled
// entirely by attn_combine (it recomputes rope/quantize itself) so the hot
// loop here has no special cases at all.
// One block per (split, b, kvh). 256 threads = 8 wave32.
__global__ __launch_bounds__(256) void attn_partial(
    const float* __restrict__ xq, const float* __restrict__ fcos,
    const float* __restrict__ fsin, const float* __restrict__ k_scaler,
    const float* __restrict__ v_scaler, const int* __restrict__ cache_k,
    const int* __restrict__ cache_v, const int* __restrict__ input_pos,
    float* __restrict__ ws)
{
  const int blk   = blockIdx.x;
  const int g     = blk % NKVH;
  const int b     = (blk / NKVH) % NB;
  const int split = blk / (NKVH * NB);
  const int tid   = threadIdx.x;
  const int wave  = tid >> 5;
  const int lane  = tid & 31;
  const int ln16  = lane & 15;
  const bool hiHalf = lane >= 16;

  __shared__ __align__(16) float qf[NREP][ND];            // RoPE'd q rows (f32)
  __shared__ __align__(16) float pld[NWAVE][TILE][4];     // per-wave P*v_scaler
  __shared__ __align__(16) float accl[NWAVE][NREP][ND];   // per-wave out partials
  __shared__ float wml[NWAVE][NREP], wll[NWAVE][NREP];

  const int P = input_pos[b];                             // cached positions: [0, P)

  // ---------------- Phase 0: RoPE q (one pair per thread) ----------------
  {
    const int r = tid >> 6;          // 0..3
    const int i = tid & 63;          // 0..63 (rotary pair)
    const float c = fcos[b * 64 + i];
    const float s = fsin[b * 64 + i];
    const float* qp = xq + ((size_t)b * NH + (g * NREP + r)) * ND;
    const float xe = qp[2 * i], xo = qp[2 * i + 1];
    qf[r][2 * i]     = xe * c - xo * s;
    qf[r][2 * i + 1] = xe * s + xo * c;
  }
  __syncthreads();

  // ---------------- Build Q A-fragments (hi/lo f16 split) ----------------
  // A 16x32 f16 layout: lane = row M (rows>=4 zero); lanes 0-15 hold K {0..7,16..23},
  // lanes 16-31 hold K {8..15,24..31}; elem e<8 -> K=kb+e, e>=8 -> K=kb+16+(e-8).
  v16h ahi[4], alo[4];
  {
    const int row = ln16;
    const int kb  = hiHalf ? 8 : 0;
#pragma unroll
    for (int c = 0; c < 4; c++) {
#pragma unroll
      for (int e = 0; e < 16; e++) {
        const int K = (e < 8) ? (kb + e) : (kb + 16 + (e - 8));
        const float v = (row < NREP) ? qf[row][c * 32 + K] : 0.f;
        const _Float16 h = (_Float16)v;
        ahi[c][e] = h;
        alo[c][e] = (_Float16)(v - (float)h);
      }
    }
  }

  const size_t bg = (size_t)b * NKVH + g;
  const int*   kc  = cache_k  + bg * NCACHE * ND;
  const int*   vc  = cache_v  + bg * NCACHE * ND;
  const float* ksc = k_scaler + bg * NCACHE;
  const float* vsc = v_scaler + bg * NCACHE;

  // ---------------- Phase 1: flash-decode over this split's tiles ----------------
  float m_r[NREP], l_r[NREP], acc[NREP][4];
#pragma unroll
  for (int r = 0; r < NREP; r++) {
    m_r[r] = -3e38f; l_r[r] = 0.f;
    acc[r][0] = acc[r][1] = acc[r][2] = acc[r][3] = 0.f;
  }
  const int dbase = lane * 4;              // this lane's 4 output dims

  const int n_tiles = (P + TILE - 1) / TILE;
  const int chunk   = (n_tiles + NSPLIT - 1) / NSPLIT;
  const int t0 = split * chunk;
  const int t1 = (t0 + chunk < n_tiles) ? (t0 + chunk) : n_tiles;

  for (int t = t0 + wave; t < t1; t += NWAVE) {
    const int base = t * TILE;

    // ---- prefetch next tile's K and V rows (counter-free latency hiding) ----
    if (t + NWAVE < t1) {
      const int basen = (t + NWAVE) * TILE;
      int sK = basen + ln16;  if (sK >= P) sK = P - 1;
      const char* kpre = (const char*)(kc + (size_t)sK * ND) + (hiHalf ? 64 : 0);
#pragma unroll
      for (int c = 0; c < 4; c++) __builtin_prefetch(kpre + c * 128, 0, 3);
      int sV = basen + lane;  if (sV >= P) sV = P - 1;
      const char* vpre = (const char*)(vc + (size_t)sV * ND);
#pragma unroll
      for (int j = 0; j < 4; j++) __builtin_prefetch(vpre + j * 128, 0, 3);
    }

    v8f cst[2];
    // ---- scores: two 16-wide subtiles, K=128 as 4 chunks of 32, hi+lo q ----
#pragma unroll
    for (int st = 0; st < 2; st++) {
      const int s  = base + st * 16 + ln16;          // this lane's column position
      const int sr = (s < P) ? s : (P - 1);          // clamp for safe reads
      const int* rowp = kc + (size_t)sr * ND + (hiHalf ? 8 : 0);

      // Issue ALL 16 B128 loads for this subtile before any use (MLP).
      v4i raw[16];
#pragma unroll
      for (int c = 0; c < 4; c++) {
        const v4i* p4 = (const v4i*)(rowp + c * 32); // K {kb..kb+7, kb+16..kb+23} per lane
#pragma unroll
        for (int q = 0; q < 4; q++) raw[c * 4 + q] = __builtin_nontemporal_load(p4 + (q & 1) + (q >> 1) * 4);
      }

      v8f cc = {};
#pragma unroll
      for (int c = 0; c < 4; c++) {
        v16h bf;
#pragma unroll
        for (int q = 0; q < 4; q++) {
          const v4i wv = raw[c * 4 + q];
          bf[4*q+0] = (_Float16)(float)wv.x; bf[4*q+1] = (_Float16)(float)wv.y;
          bf[4*q+2] = (_Float16)(float)wv.z; bf[4*q+3] = (_Float16)(float)wv.w;
        }
        cc = __builtin_amdgcn_wmma_f32_16x16x32_f16(false, ahi[c], false, bf, (short)0, cc, false, false);
        cc = __builtin_amdgcn_wmma_f32_16x16x32_f16(false, alo[c], false, bf, (short)0, cc, false, false);
      }
      cst[st] = cc;
    }

    // ---- scale by k_scaler/sqrt(D), mask, broadcast rows 0..3 wave-wide ----
    const int s0 = base + ln16,         s1 = base + 16 + ln16;
    const int s0r = (s0 < P) ? s0 : P-1, s1r = (s1 < P) ? s1 : P-1;
    const float kscl0 = ksc[s0r] * INV_SQRT_D;
    const float kscl1 = ksc[s1r] * INV_SQRT_D;
    const float vscl0 = vsc[s0r];
    const float vscl1 = vsc[s1r];
    float sc0[NREP], sc1[NREP];
#pragma unroll
    for (int r = 0; r < NREP; r++) {
      const float a = __shfl(cst[0][r], ln16, 32);   // rows 0..3 live in lanes 0..15
      const float c = __shfl(cst[1][r], ln16, 32);
      sc0[r] = (s0 < P) ? a * kscl0 : -1e30f;
      sc1[r] = (s1 < P) ? c * kscl1 : -1e30f;
    }

    // ---- online softmax update (state wave-uniform) ----
#pragma unroll
    for (int r = 0; r < NREP; r++) {
      const float tm    = grp16_max(fmaxf(sc0[r], sc1[r]));
      const float mnew  = fmaxf(m_r[r], tm);
      const float alpha = __expf(m_r[r] - mnew);
      const float p0 = __expf(sc0[r] - mnew);
      const float p1 = __expf(sc1[r] - mnew);
      l_r[r] = l_r[r] * alpha + grp16_sum(p0 + p1);
      m_r[r] = mnew;
      acc[r][0] *= alpha; acc[r][1] *= alpha; acc[r][2] *= alpha; acc[r][3] *= alpha;
      if (!hiHalf) {
        pld[wave][ln16][r]      = p0 * vscl0;        // fold v_scaler into P
        pld[wave][16 + ln16][r] = p1 * vscl1;
      }
    }

    // ---- P.V: d-per-lane, full 512B coalesced row per iteration ----
    const int nvalid = ((P - base) < TILE) ? (P - base) : TILE;
#pragma unroll 8
    for (int si = 0; si < nvalid; si++) {
      const int s = base + si;
      const v4f pv = *(const v4f*)&pld[wave][si][0];
      const v4i wv = __builtin_nontemporal_load((const v4i*)(vc + (size_t)s * ND + dbase));
      const float f0 = (float)wv.x, f1 = (float)wv.y, f2 = (float)wv.z, f3 = (float)wv.w;
#pragma unroll
      for (int r = 0; r < NREP; r++) {
        const float p = pv[r];
        acc[r][0] = fmaf(p, f0, acc[r][0]);
        acc[r][1] = fmaf(p, f1, acc[r][1]);
        acc[r][2] = fmaf(p, f2, acc[r][2]);
        acc[r][3] = fmaf(p, f3, acc[r][3]);
      }
    }
  }

  // ---------------- cross-wave combine in LDS, emit one partial ----------------
#pragma unroll
  for (int r = 0; r < NREP; r++) {
    v4f av; av[0] = acc[r][0]; av[1] = acc[r][1]; av[2] = acc[r][2]; av[3] = acc[r][3];
    *(v4f*)&accl[wave][r][dbase] = av;
    if (lane == 0) { wml[wave][r] = m_r[r]; wll[wave][r] = l_r[r]; }
  }
  __syncthreads();

  float* wout = ws + ((size_t)split * NB * NKVH + bg) * PSZ;
  for (int idx = tid; idx < NREP * ND; idx += 256) {
    const int r = idx / ND, d = idx % ND;
    float M = -3e38f;
#pragma unroll
    for (int w = 0; w < NWAVE; w++) M = fmaxf(M, wml[w][r]);
    float o = 0.f, L = 0.f;
#pragma unroll
    for (int w = 0; w < NWAVE; w++) {
      const float e = __expf(wml[w][r] - M);        // empty wave: exp(-inf)=0
      o += accl[w][r][d] * e;
      L += wll[w][r] * e;
    }
    wout[idx] = o;
    if (d == 0) { wout[NREP * ND + r] = M; wout[NREP * ND + NREP + r] = L; }
  }
}

// Merges NSPLIT flash partials AND the new token (s == P) column, which it
// recomputes from the raw inputs (RoPE + int8 quantization, exactly as the
// reference inserts into the cache).  One block of 128 threads per (b, kvh).
__global__ __launch_bounds__(128) void attn_combine(
    const float* __restrict__ xq, const float* __restrict__ xk,
    const float* __restrict__ xv, const float* __restrict__ fcos,
    const float* __restrict__ fsin, const int* __restrict__ input_pos,
    const float* __restrict__ ws, float* __restrict__ out)
{
  const int blk = blockIdx.x;                        // b*NKVH + g
  const int g = blk % NKVH, b = blk / NKVH;
  const int tid = threadIdx.x;                       // 0..127

  __shared__ __align__(16) float qr[NREP][ND];       // RoPE'd q
  __shared__ __align__(16) float kq[ND];             // new k: rope'd then quantized
  __shared__ __align__(16) float vq[ND];             // new v quantized
  __shared__ float red[2];                           // k_s_new, v_s_new
  __shared__ float snew[NREP];                       // new-token scores

  if (tid < 64) {                                    // RoPE pairs
    const int i = tid;
    const float c = fcos[b * 64 + i], s = fsin[b * 64 + i];
#pragma unroll
    for (int r = 0; r < NREP; r++) {
      const float* qp = xq + ((size_t)b * NH + g * NREP + r) * ND;
      const float xe = qp[2 * i], xo = qp[2 * i + 1];
      qr[r][2 * i] = xe * c - xo * s;  qr[r][2 * i + 1] = xe * s + xo * c;
    }
    const float* kp = xk + ((size_t)b * NKVH + g) * ND;
    const float ke = kp[2 * i], ko = kp[2 * i + 1];
    kq[2 * i] = ke * c - ko * s;  kq[2 * i + 1] = ke * s + ko * c;
  }
  vq[tid] = xv[((size_t)b * NKVH + g) * ND + tid];
  __syncthreads();

  if (tid < 32) {                                    // abs-max over 128
    float mk = 0.f, mv = 0.f;
#pragma unroll
    for (int j = 0; j < 4; j++) {
      mk = fmaxf(mk, fabsf(kq[tid + 32 * j]));
      mv = fmaxf(mv, fabsf(vq[tid + 32 * j]));
    }
#pragma unroll
    for (int off = 1; off < 32; off <<= 1) {
      mk = fmaxf(mk, __shfl_xor(mk, off, 32));
      mv = fmaxf(mv, __shfl_xor(mv, off, 32));
    }
    if (tid == 0) { red[0] = mk / 127.f + 1e-8f; red[1] = mv / 127.f + 1e-8f; }
  }
  __syncthreads();
  const float ksn = red[0], vsn = red[1];
  {                                                  // quantize (RNE like jnp.round)
    const float k2 = rintf(kq[tid] / ksn); kq[tid] = fminf(fmaxf(k2, -127.f), 127.f);
    const float v2 = rintf(vq[tid] / vsn); vq[tid] = fminf(fmaxf(v2, -127.f), 127.f);
  }
  __syncthreads();
  if (tid < 32) {                                    // new-token scores (4 dots of 128)
#pragma unroll
    for (int r = 0; r < NREP; r++) {
      float p = 0.f;
#pragma unroll
      for (int j = 0; j < 4; j++) p += qr[r][tid * 4 + j] * kq[tid * 4 + j];
#pragma unroll
      for (int off = 1; off < 32; off <<= 1) p += __shfl_xor(p, off, 32);
      if (tid == 0) snew[r] = p * INV_SQRT_D * ksn;
    }
  }
  __syncthreads();

  const int d = tid;
#pragma unroll
  for (int r = 0; r < NREP; r++) {
    float M = snew[r];                               // new token always valid
#pragma unroll
    for (int p = 0; p < NSPLIT; p++)
      M = fmaxf(M, ws[((size_t)p * NB * NKVH + blk) * PSZ + NREP * ND + r]);
    float en = __expf(snew[r] - M);
    float L = en;
    float o = en * vsn * vq[d];                      // new token's P*v_scaler*V
#pragma unroll
    for (int p = 0; p < NSPLIT; p++) {
      const float* wp = ws + ((size_t)p * NB * NKVH + blk) * PSZ;
      const float e = __expf(wp[NREP * ND + r] - M);
      L += wp[NREP * ND + NREP + r] * e;
      o += wp[(size_t)r * ND + d] * e;
    }
    out[((size_t)b * NH + g * NREP + r) * ND + d] = o / L;
  }
}

extern "C" void kernel_launch(void* const* d_in, const int* in_sizes, int n_in,
                              void* d_out, int out_size, void* d_ws, size_t ws_size,
                              hipStream_t stream)
{
  const float* xq        = (const float*)d_in[0];
  const float* xk        = (const float*)d_in[1];
  const float* xv        = (const float*)d_in[2];
  const float* fcos      = (const float*)d_in[3];
  const float* fsin      = (const float*)d_in[4];
  const float* k_scaler  = (const float*)d_in[5];
  const float* v_scaler  = (const float*)d_in[6];
  const int*   cache_k   = (const int*)d_in[7];
  const int*   cache_v   = (const int*)d_in[8];
  const int*   input_pos = (const int*)d_in[9];
  float* ws  = (float*)d_ws;
  float* out = (float*)d_out;

  attn_partial<<<dim3(NB * NKVH * NSPLIT), dim3(256), 0, stream>>>(
      xq, fcos, fsin, k_scaler, v_scaler, cache_k, cache_v, input_pos, ws);
  attn_combine<<<dim3(NB * NKVH), dim3(128), 0, stream>>>(
      xq, xk, xv, fcos, fsin, input_pos, ws, out);
}
